// GraphCL_82317343195923
// MI455X (gfx1250) — matrix operand order repaired and checked by reading, not compiled, not run on hardware
//
#include <hip/hip_runtime.h>
#include <hip/hip_bf16.h>

typedef __attribute__((ext_vector_type(16))) __bf16 v16bf;
typedef __attribute__((ext_vector_type(8)))  __bf16 v8bf;
typedef __attribute__((ext_vector_type(8)))  float  v8f;

__device__ __forceinline__ __bf16 f2bf(float f) {
    unsigned u = __builtin_bit_cast(unsigned, f);
    unsigned r = u + 0x7FFFu + ((u >> 16) & 1u);   // round-to-nearest-even
    unsigned short s = (unsigned short)(r >> 16);
    return __builtin_bit_cast(__bf16, s);
}

// ---------------------------------------------------------------------------
// Zero a float buffer
// ---------------------------------------------------------------------------
__global__ void zero_kernel(float* __restrict__ p, size_t n) {
    size_t i = (size_t)blockIdx.x * blockDim.x + threadIdx.x;
    if (i < n) p[i] = 0.0f;
}

// ---------------------------------------------------------------------------
// f32 -> bf16 bulk conversion (n multiple of 4)
// ---------------------------------------------------------------------------
__global__ void cvt_bf16_kernel(const float* __restrict__ s, __bf16* __restrict__ d, size_t n4) {
    size_t i = (size_t)blockIdx.x * blockDim.x + threadIdx.x;
    if (i >= n4) return;
    float4 v = ((const float4*)s)[i];
    __bf16* dp = d + i * 4;
    dp[0] = f2bf(v.x); dp[1] = f2bf(v.y); dp[2] = f2bf(v.z); dp[3] = f2bf(v.w);
}

// ---------------------------------------------------------------------------
// Degree accumulation over edge destinations (hardware f32 atomics into L2)
// ---------------------------------------------------------------------------
__global__ void deg_kernel(const int* __restrict__ dst, float* __restrict__ deg, int E) {
    int e = blockIdx.x * blockDim.x + threadIdx.x;
    if (e < E)
        __hip_atomic_fetch_add(&deg[dst[e]], 1.0f, __ATOMIC_RELAXED, __HIP_MEMORY_SCOPE_AGENT);
}

// dinv = rsqrt(deg + 1)   (+1 = self loop, so always > 0), in place
__global__ void dinv_kernel(float* __restrict__ deg, int N) {
    int n = blockIdx.x * blockDim.x + threadIdx.x;
    if (n < N) deg[n] = rsqrtf(deg[n] + 1.0f);
}

// ---------------------------------------------------------------------------
// WMMA GEMM:  C[M,Nc] = A[M,K] * W[Nc,K]^T (+bias, optional ReLU), Nc = 128.
// A, W bf16; C f32 or bf16. One wave computes a 16x64 tile (4 accumulators
// share one A fragment). The weight matrix is staged in LDS in 128-wide K
// slabs (32KB + 16B/row pad to rotate banks); A streams from global.
// Per k-step: 2 A global_load_b128 + 8 B ds_load_b128 + 4 WMMAs.
// ---------------------------------------------------------------------------
template<int K, bool RELU, bool OUTBF>
__global__ __launch_bounds__(256)
void wmma_gemm_kernel(const __bf16* __restrict__ A, int lda,
                      const __bf16* __restrict__ W,
                      const float* __restrict__ bias,
                      void* __restrict__ Cout, int ldc,
                      int Mtiles, int NWcols)       // NWcols = Nc/64 = 2
{
    constexpr int SROW = 136;                       // 128 + 8 bf16 pad (16B)
    __shared__ __bf16 sW[128 * SROW];

    int wave = blockIdx.x * (blockDim.x >> 5) + (threadIdx.x >> 5);
    bool active = wave < Mtiles * NWcols;           // wave-uniform
    int tm = wave / NWcols;
    int tc = wave - tm * NWcols;

    int lane = threadIdx.x & 31;
    int lo = lane & 15;        // A row within tile, B/C column within tile
    int hi = lane >> 4;        // lane-half selector

    const __bf16* arow = active ? A + (size_t)(tm * 16 + lo) * lda : A;
    const __bf16* srow = sW + (size_t)(tc * 64 + lo) * SROW;

    v8f c0 = {}, c1 = {}, c2 = {}, c3 = {};
#pragma unroll
    for (int kk = 0; kk < K; kk += 128) {
        if (kk) __syncthreads();                    // sW reuse across slabs
        // cooperative stage: 128 rows x 128 k of W into LDS (b128 granules)
        for (int i = threadIdx.x; i < 128 * 16; i += 256) {
            int n = i >> 4, cchunk = i & 15;
            *(v8bf*)(sW + n * SROW + cchunk * 8) =
                *(const v8bf*)(W + (size_t)n * K + kk + cchunk * 8);
        }
        __syncthreads();

        if (active) {
            v8f* cs[4] = {&c0, &c1, &c2, &c3};
#pragma unroll
            for (int k0 = 0; k0 < 128; k0 += 32) {
                // A 16x32 bf16 fragment: two 16B loads
                v8bf al = *(const v8bf*)(arow + kk + k0 + hi * 8);
                v8bf ah = *(const v8bf*)(arow + kk + k0 + 16 + hi * 8);
                v16bf a = __builtin_shufflevector(al, ah,
                              0,1,2,3,4,5,6,7,8,9,10,11,12,13,14,15);
#pragma unroll
                for (int t = 0; t < 4; ++t) {
                    const __bf16* bp = srow + (size_t)t * 16 * SROW + k0 + hi * 8;
                    v8bf bl = *(const v8bf*)(bp);
                    v8bf bh = *(const v8bf*)(bp + 16);
                    v16bf b = __builtin_shufflevector(bl, bh,
                                  0,1,2,3,4,5,6,7,8,9,10,11,12,13,14,15);
                    *cs[t] = __builtin_amdgcn_wmma_f32_16x16x32_bf16(
                                 false, a, false, b, (short)0, *cs[t], false, false);
                }
            }
        }
    }

    if (!active) return;
    v8f cc[4] = {c0, c1, c2, c3};
#pragma unroll
    for (int t = 0; t < 4; ++t) {
        int col = tc * 64 + t * 16 + lo;
        float bs = bias ? bias[col] : 0.0f;
#pragma unroll
        for (int r = 0; r < 8; ++r) {  // C/D layout: VGPR r -> row r (lanes 0-15) / r+8
            int row = tm * 16 + r + hi * 8;
            float v = cc[t][r] + bs;
            if (RELU) v = fmaxf(v, 0.0f);
            if (OUTBF) ((__bf16*)Cout)[(size_t)row * ldc + col] = f2bf(v);
            else       ((float*) Cout)[(size_t)row * ldc + col] = v;
        }
    }
}

// ---------------------------------------------------------------------------
// Edge aggregation: one wave per edge, lane l moves float4 l of the 128-f32 row.
// acc[dst] += h[src] * dinv[src]*dinv[dst]   via hardware f32 atomics (L2-resident)
// ---------------------------------------------------------------------------
__global__ __launch_bounds__(256)
void edge_scatter_kernel(const float* __restrict__ h, const float* __restrict__ dinv,
                         const int* __restrict__ src, const int* __restrict__ dst,
                         float* __restrict__ acc, int E)
{
    int gtid = blockIdx.x * blockDim.x + threadIdx.x;
    int e = gtid >> 5;
    if (e >= E) return;               // wave-uniform exit
    int lane = threadIdx.x & 31;

    int s = src[e];
    int d = dst[e];
    float nrm = dinv[s] * dinv[d];

    float4 hv = ((const float4*)(h + (size_t)s * 128))[lane];
    float* a = acc + (size_t)d * 128 + lane * 4;
    __hip_atomic_fetch_add(a + 0, hv.x * nrm, __ATOMIC_RELAXED, __HIP_MEMORY_SCOPE_AGENT);
    __hip_atomic_fetch_add(a + 1, hv.y * nrm, __ATOMIC_RELAXED, __HIP_MEMORY_SCOPE_AGENT);
    __hip_atomic_fetch_add(a + 2, hv.z * nrm, __ATOMIC_RELAXED, __HIP_MEMORY_SCOPE_AGENT);
    __hip_atomic_fetch_add(a + 3, hv.w * nrm, __ATOMIC_RELAXED, __HIP_MEMORY_SCOPE_AGENT);
}

// ---------------------------------------------------------------------------
// Layer finalize: v = relu(acc + h*dinv^2 (self loop) + bias); written as bf16
// into the layer's 128-wide column slab of the [N,384] bf16 concat buffer.
// ---------------------------------------------------------------------------
__global__ void finalize_kernel(const float* __restrict__ acc, const float* __restrict__ h,
                                const float* __restrict__ dinv, const float* __restrict__ b,
                                __bf16* __restrict__ hcat_col, size_t total)
{
    size_t i = (size_t)blockIdx.x * blockDim.x + threadIdx.x;
    if (i >= total) return;
    int n  = (int)(i >> 7);
    int ch = (int)(i & 127);
    float di = dinv[n];
    float v = acc[i] + h[i] * di * di + b[ch];
    v = fmaxf(v, 0.0f);
    hcat_col[(size_t)n * 384 + ch] = f2bf(v);
}

// ---------------------------------------------------------------------------

extern "C" void kernel_launch(void* const* d_in, const int* in_sizes, int n_in,
                              void* d_out, int out_size, void* d_ws, size_t ws_size,
                              hipStream_t stream)
{
    const float* x   = (const float*)d_in[0];
    const int*   ei  = (const int*)  d_in[1];
    const float* W0  = (const float*)d_in[2];
    const float* b0  = (const float*)d_in[3];
    const float* W1  = (const float*)d_in[4];
    const float* b1  = (const float*)d_in[5];
    const float* W2  = (const float*)d_in[6];
    const float* b2  = (const float*)d_in[7];
    const float* Wp0 = (const float*)d_in[8];
    const float* bp0 = (const float*)d_in[9];
    const float* Wp1 = (const float*)d_in[10];
    const float* bp1 = (const float*)d_in[11];

    const int N = in_sizes[0] / 128;     // 100000
    const int E = in_sizes[1] / 2;       // 1600000
    const int* src = ei;
    const int* dst = ei + E;

    // ---- workspace carve-up ----
    auto al512 = [](size_t x) { return (x + 511) / 512 * 512; };
    char* w = (char*)d_ws;
    float*  dinv = (float*)w;  w += al512((size_t)N * 4);
    float*  h    = (float*)w;  w += (size_t)N * 128 * 4;
    float*  acc  = (float*)w;  w += (size_t)N * 128 * 4;
    __bf16* xb   = (__bf16*)w; w += al512((size_t)N * 128 * 2);
    __bf16* hcb  = (__bf16*)w; w += al512((size_t)N * 384 * 2);
    __bf16* zb   = (__bf16*)w; w += al512((size_t)N * 128 * 2);
    __bf16* w0b  = (__bf16*)w; w += (size_t)128 * 128 * 2;
    __bf16* w1b  = (__bf16*)w; w += (size_t)128 * 128 * 2;
    __bf16* w2b  = (__bf16*)w; w += (size_t)128 * 128 * 2;
    __bf16* wp0b = (__bf16*)w; w += (size_t)128 * 384 * 2;
    __bf16* wp1b = (__bf16*)w; w += (size_t)128 * 128 * 2;

    const size_t NF = (size_t)N * 128;
    const int B = 256;
    int nBlkNode = (N + B - 1) / B;
    int nBlkEdge = (E + B - 1) / B;
    int nBlkFeat = (int)((NF + B - 1) / B);
    int nBlkEdgeWave = (E * 32 + B - 1) / B;    // one wave per edge

    auto cvt = [&](const float* s, __bf16* d, size_t n) {
        size_t n4 = n / 4;
        cvt_bf16_kernel<<<(int)((n4 + B - 1) / B), B, 0, stream>>>(s, d, n4);
    };

    // ---- one-time bf16 conversions (weights + input features) ----
    cvt(x,   xb,   NF);
    cvt(W0,  w0b,  (size_t)128 * 128);
    cvt(W1,  w1b,  (size_t)128 * 128);
    cvt(W2,  w2b,  (size_t)128 * 128);
    cvt(Wp0, wp0b, (size_t)128 * 384);
    cvt(Wp1, wp1b, (size_t)128 * 128);

    // ---- degrees: deg = #incoming edges; dinv = rsqrt(deg+1) ----
    zero_kernel<<<nBlkNode, B, 0, stream>>>(dinv, (size_t)N);
    deg_kernel<<<nBlkEdge, B, 0, stream>>>(dst, dinv, E);
    dinv_kernel<<<nBlkNode, B, 0, stream>>>(dinv, N);

    // ---- GEMM launch geometry: 16x64 per wave, Nc = 128 -> 2 wave-cols ----
    const int Mtiles = N / 16;                 // 6250
    const int NWcols = 128 / 64;               // 2
    const int gemmBlocks = (Mtiles * NWcols + 7) / 8;

    // ---- 3 GCN layers ----
    const __bf16* layerWb[3] = {w0b, w1b, w2b};
    const float*  layerB[3]  = {b0, b1, b2};
    for (int l = 0; l < 3; ++l) {
        const __bf16* inA = (l == 0) ? xb : (hcb + (size_t)(l - 1) * 128);
        int lda           = (l == 0) ? 128 : 384;
        wmma_gemm_kernel<128, false, false><<<gemmBlocks, B, 0, stream>>>(
            inA, lda, layerWb[l], nullptr, h, 128, Mtiles, NWcols);
        zero_kernel<<<nBlkFeat, B, 0, stream>>>(acc, NF);
        edge_scatter_kernel<<<nBlkEdgeWave, B, 0, stream>>>(h, dinv, src, dst, acc, E);
        finalize_kernel<<<nBlkFeat, B, 0, stream>>>(acc, h, dinv, layerB[l],
                                                    hcb + (size_t)l * 128, NF);
    }

    // ---- projection head ----
    wmma_gemm_kernel<384, true, true><<<gemmBlocks, B, 0, stream>>>(
        hcb, 384, wp0b, bp0, zb, 128, Mtiles, NWcols);
    wmma_gemm_kernel<128, false, false><<<gemmBlocks, B, 0, stream>>>(
        zb, 128, wp1b, bp1, d_out, 128, Mtiles, NWcols);
}
